// MaskSelfAttention_SquareGrowth_47768626266371
// MI455X (gfx1250) — matrix operand order repaired and checked by reading, not compiled, not run on hardware
//
#include <hip/hip_runtime.h>

typedef __attribute__((ext_vector_type(16))) __bf16 v16bf;
typedef __attribute__((ext_vector_type(8)))  float  v8f;

#define WMMA_BF16(a, b, c) \
  __builtin_amdgcn_wmma_f32_16x16x32_bf16(false, (a), false, (b), (short)0, (c), false, false)

// ---- problem constants ----
#define BATCH   2
#define SEQ     2048
#define CDIM    1024
#define NHEADS  16
#define HDIM    64
#define SCALE   0.125f          // 64^-0.5 (exact power of two -> folded into Q)
#define ROWS    (BATCH * SEQ)   // 4096

// =============== WMMA fragment loaders (ISA 7.12.2 layouts) ===============

// A fragment: 16x32 (MxK), K-contiguous per lane -> 2x ds_load_b128.
__device__ __forceinline__ v16bf load_a_frag(const __bf16* p, int stride, int lane) {
  const int m = lane & 15, half = (lane >> 4) & 1;
  v16bf f;
#pragma unroll
  for (int e = 0; e < 8; ++e) {
    f[e]     = p[m * stride + half * 8 + e];
    f[e + 8] = p[m * stride + 16 + half * 8 + e];
  }
  return f;
}

// B fragment: 32x16 (KxN) read from N-MAJOR storage (mem[n*stride + k]):
// lane n = lane%16, elem e<8 -> K = half*8 + e, e>=8 -> K = 16 + half*8 + (e-8).
__device__ __forceinline__ v16bf load_bT_frag(const __bf16* p, int stride, int lane) {
  const int n = lane & 15, half = (lane >> 4) & 1;
  v16bf f;
#pragma unroll
  for (int e = 0; e < 8; ++e) {
    f[e]     = p[n * stride + half * 8 + e];
    f[e + 8] = p[n * stride + 16 + half * 8 + e];
  }
  return f;
}

// ======================= Kernel 1: QKV projection ==========================
// One instantiation per Q/K/V segment: C_seg[4096 x 1024] = h @ w_seg.
// Block tile 64x128, 8 waves, each wave a 32x32 tile (4 wmma / k-step).
// THREE=0: Q (pre-scaled) -> [B][H][N][D]; THREE=1: K -> [B][H][N][D];
// THREE=2: V -> [B][H][D][N] (transposed for attention B-frags).
template <int THREE>
__global__ __launch_bounds__(256) void qkv_gemm_kernel(
    const float* __restrict__ h, const float* __restrict__ w,
    __bf16* __restrict__ dst) {
  __shared__ __bf16 ldsA[64][32];     // [m][k]
  __shared__ __bf16 ldsBt[128][32];   // [n][k]  (N-major!)
  const int t = threadIdx.x;
  const int wv = t >> 5, lane = t & 31;
  const int row0 = blockIdx.y * 64;
  const int col0 = blockIdx.x * 128;  // 0..1023 within this segment
  const int mb = (wv >> 2) * 32;      // 0 or 32
  const int nb = (wv & 3) * 32;       // 0..96
  v8f acc00 = {}, acc01 = {}, acc10 = {}, acc11 = {};

  for (int k0 = 0; k0 < CDIM; k0 += 32) {
    __syncthreads();
#pragma unroll
    for (int i = 0; i < 8; ++i) {            // A: 64x32, f32 -> bf16
      int e = i * 256 + t;
      int r = e >> 5, c = e & 31;
      ldsA[r][c] = (__bf16)h[(size_t)(row0 + r) * CDIM + k0 + c];
    }
#pragma unroll
    for (int i = 0; i < 16; ++i) {           // B: 32x128 -> transposed store
      int e = i * 256 + t;
      int c = e & 127, r = e >> 7;           // coalesced global read along c
      ldsBt[c][r] = (__bf16)w[(size_t)(k0 + r) * 3072 + col0 + c];
    }
    __syncthreads();
    v16bf a0 = load_a_frag(&ldsA[mb][0],      32, lane);
    v16bf a1 = load_a_frag(&ldsA[mb + 16][0], 32, lane);
    v16bf b0 = load_bT_frag(&ldsBt[nb][0],      32, lane);
    v16bf b1 = load_bT_frag(&ldsBt[nb + 16][0], 32, lane);
    acc00 = WMMA_BF16(a0, b0, acc00);
    acc01 = WMMA_BF16(a0, b1, acc01);
    acc10 = WMMA_BF16(a1, b0, acc10);
    acc11 = WMMA_BF16(a1, b1, acc11);
  }

  const int half = lane >> 4, n16 = lane & 15;
  const v8f* accs[4] = {&acc00, &acc01, &acc10, &acc11};
#pragma unroll
  for (int tt = 0; tt < 4; ++tt) {
    const v8f a = *accs[tt];
    const int tm = tt >> 1, tn = tt & 1;
#pragma unroll
    for (int j = 0; j < 8; ++j) {
      int r  = row0 + mb + tm * 16 + j + 8 * half;   // global row -> (b, n)
      int cl = col0 + nb + tn * 16 + n16;            // segment col -> (hh, d)
      int hh = cl >> 6;
      int d  = cl & 63;
      int b  = r >> 11;
      int n  = r & (SEQ - 1);
      if constexpr (THREE == 2) {   // V: [B][H][D][N]
        size_t off = (((size_t)(b * NHEADS + hh)) * HDIM + d) * SEQ + n;
        dst[off] = (__bf16)a[j];
      } else {                      // Q/K: [B][H][N][D]
        size_t off = (((size_t)(b * NHEADS + hh)) * SEQ + n) * HDIM + d;
        dst[off] = (__bf16)(THREE == 0 ? a[j] * SCALE : a[j]);
      }
    }
  }
}

// ==================== Kernel 2: masked flash attention =====================
// Wave = one 16-query tile; block = 8 waves of one (b,h) sharing 32-key
// K/V tiles staged via GLOBAL_LOAD_ASYNC_TO_LDS_B128 (ASYNCcnt).
__global__ __launch_bounds__(256) void attn_kernel(
    const __bf16* __restrict__ qkv, const float* __restrict__ mask,
    __bf16* __restrict__ attn_out) {
  __shared__ __bf16 ldsK[32][64];        // [key][d]  (K^T frags contiguous)
  __shared__ __bf16 ldsVt[64][32];       // [d][key]  (V frags contiguous)
  __shared__ __bf16 ldsP[8][16][32];     // wave-private P reshape

  const int t = threadIdx.x, wv = t >> 5, lane = t & 31;
  const int half = lane >> 4, n16 = lane & 15;
  const int blk  = blockIdx.x;        // 512 blocks
  const int qblk = blk & 15;
  const int bh   = blk >> 4;          // b*16 + h
  const int b    = bh >> 4;

  const size_t BHND = (size_t)BATCH * NHEADS * SEQ * HDIM;
  const __bf16* kbase = qkv + BHND + (size_t)bh * SEQ * HDIM;      // [n][d]
  const __bf16* vbase = qkv + 2 * BHND + (size_t)bh * HDIM * SEQ;  // [d][n]

  const int qtile = qblk * 8 + wv;
  const __bf16* qp = qkv + (size_t)bh * SEQ * HDIM + (size_t)qtile * 16 * HDIM;
  v16bf qa0 = load_a_frag(qp, HDIM, lane);        // d = 0..31 (Q pre-scaled)
  v16bf qa1 = load_a_frag(qp + 32, HDIM, lane);   // d = 32..63

  const unsigned ldsK_off = (unsigned)(size_t)&ldsK[0][0];
  const unsigned ldsV_off = (unsigned)(size_t)&ldsVt[0][0];

  float mrow[8], lrow[8];
  v8f o0 = {}, o1 = {}, o2 = {}, o3 = {};
#pragma unroll
  for (int j = 0; j < 8; ++j) { mrow[j] = -__builtin_inff(); lrow[j] = 0.f; }

  for (int kt = 0; kt < SEQ / 32; ++kt) {
    __syncthreads();
    {
      // K tile: 4KB contiguous; 8 waves x 32 lanes x 16B = 4KB (1 instr/wave)
      unsigned kdst  = ldsK_off + (unsigned)(wv * 512 + lane * 16);
      unsigned kvoff = (unsigned)(kt * 4096 + wv * 512 + lane * 16);
      asm volatile("global_load_async_to_lds_b128 %0, %1, %2"
                   :: "v"(kdst), "v"(kvoff), "s"(kbase) : "memory");
      // V tile: 64 rows (d) x 64B from a 4096B-stride matrix
      unsigned vdst  = ldsV_off + (unsigned)(wv * 512 + lane * 16);
      unsigned vvoff = (unsigned)((wv * 8 + (lane >> 2)) * (SEQ * 2) +
                                  kt * 64 + (lane & 3) * 16);
      asm volatile("global_load_async_to_lds_b128 %0, %1, %2"
                   :: "v"(vdst), "v"(vvoff), "s"(vbase) : "memory");
      asm volatile("s_wait_asynccnt 0" ::: "memory");
    }
    __syncthreads();

    // S = Q @ K^T  (16 x 32), f32 accum; K^T frag = contiguous row of ldsK
    v8f s0 = {}, s1 = {};
    s0 = WMMA_BF16(qa0, load_bT_frag(&ldsK[0][0],   HDIM, lane), s0);
    s0 = WMMA_BF16(qa1, load_bT_frag(&ldsK[0][32],  HDIM, lane), s0);
    s1 = WMMA_BF16(qa0, load_bT_frag(&ldsK[16][0],  HDIM, lane), s1);
    s1 = WMMA_BF16(qa1, load_bT_frag(&ldsK[16][32], HDIM, lane), s1);

    const float mk0 = mask[b * SEQ + kt * 32 + n16];
    const float mk1 = mask[b * SEQ + kt * 32 + 16 + n16];

#pragma unroll
    for (int j = 0; j < 8; ++j) {
      float x0 = s0[j], x1 = s1[j];      // already scaled (Q pre-scaled)
      float tm = fmaxf(x0, x1);
      tm = fmaxf(tm, __shfl_xor(tm, 1));
      tm = fmaxf(tm, __shfl_xor(tm, 2));
      tm = fmaxf(tm, __shfl_xor(tm, 4));
      tm = fmaxf(tm, __shfl_xor(tm, 8));
      float mn   = fmaxf(mrow[j], tm);
      float corr = __expf(mrow[j] - mn);
      float p0 = __expf(x0 - mn);
      float p1 = __expf(x1 - mn);
      float rs = p0 + p1;               // UNMASKED denom (matches reference)
      rs += __shfl_xor(rs, 1);
      rs += __shfl_xor(rs, 2);
      rs += __shfl_xor(rs, 4);
      rs += __shfl_xor(rs, 8);
      lrow[j] = lrow[j] * corr + rs;
      mrow[j] = mn;
      o0[j] *= corr; o1[j] *= corr; o2[j] *= corr; o3[j] *= corr;
      ldsP[wv][j + 8 * half][n16]      = (__bf16)(p0 * mk0);
      ldsP[wv][j + 8 * half][16 + n16] = (__bf16)(p1 * mk1);
    }

    // O += P @ V : V frags contiguous from ldsVt[d][key]
    v16bf pa = load_a_frag(&ldsP[wv][0][0], 32, lane);
    o0 = WMMA_BF16(pa, load_bT_frag(&ldsVt[0][0],  32, lane), o0);
    o1 = WMMA_BF16(pa, load_bT_frag(&ldsVt[16][0], 32, lane), o1);
    o2 = WMMA_BF16(pa, load_bT_frag(&ldsVt[32][0], 32, lane), o2);
    o3 = WMMA_BF16(pa, load_bT_frag(&ldsVt[48][0], 32, lane), o3);
  }

  // finalize: divide by denom, store [B][N][C], col = h*64 + d (bf16)
#pragma unroll
  for (int j = 0; j < 8; ++j) {
    float inv = 1.0f / lrow[j];
    int q = qtile * 16 + j + 8 * half;
    size_t base = ((size_t)b * SEQ + q) * CDIM + (bh & 15) * HDIM;
    attn_out[base + 0  + n16] = (__bf16)(o0[j] * inv);
    attn_out[base + 16 + n16] = (__bf16)(o1[j] * inv);
    attn_out[base + 32 + n16] = (__bf16)(o2[j] * inv);
    attn_out[base + 48 + n16] = (__bf16)(o3[j] * inv);
  }
}

// ================== Kernel 3: output projection + bias =====================
__global__ __launch_bounds__(256) void proj_gemm_kernel(
    const __bf16* __restrict__ a, const float* __restrict__ w,
    const float* __restrict__ bias, float* __restrict__ out) {
  __shared__ __bf16 ldsA[64][32];
  __shared__ __bf16 ldsBt[128][32];
  const int t = threadIdx.x;
  const int wv = t >> 5, lane = t & 31;
  const int row0 = blockIdx.y * 64;
  const int col0 = blockIdx.x * 128;
  const int mb = (wv >> 2) * 32;
  const int nb = (wv & 3) * 32;
  v8f acc00 = {}, acc01 = {}, acc10 = {}, acc11 = {};

  for (int k0 = 0; k0 < CDIM; k0 += 32) {
    __syncthreads();
#pragma unroll
    for (int i = 0; i < 8; ++i) {
      int e = i * 256 + t;
      int r = e >> 5, c = e & 31;
      ldsA[r][c] = a[(size_t)(row0 + r) * CDIM + k0 + c];
    }
#pragma unroll
    for (int i = 0; i < 16; ++i) {
      int e = i * 256 + t;
      int c = e & 127, r = e >> 7;
      ldsBt[c][r] = (__bf16)w[(size_t)(k0 + r) * CDIM + col0 + c];
    }
    __syncthreads();
    v16bf a0 = load_a_frag(&ldsA[mb][0],      32, lane);
    v16bf a1 = load_a_frag(&ldsA[mb + 16][0], 32, lane);
    v16bf b0 = load_bT_frag(&ldsBt[nb][0],      32, lane);
    v16bf b1 = load_bT_frag(&ldsBt[nb + 16][0], 32, lane);
    acc00 = WMMA_BF16(a0, b0, acc00);
    acc01 = WMMA_BF16(a0, b1, acc01);
    acc10 = WMMA_BF16(a1, b0, acc10);
    acc11 = WMMA_BF16(a1, b1, acc11);
  }

  const int half = lane >> 4, n16 = lane & 15;
  const v8f* accs[4] = {&acc00, &acc01, &acc10, &acc11};
#pragma unroll
  for (int tt = 0; tt < 4; ++tt) {
    const v8f av = *accs[tt];
    const int tm = tt >> 1, tn = tt & 1;
#pragma unroll
    for (int j = 0; j < 8; ++j) {
      int r  = row0 + mb + tm * 16 + j + 8 * half;
      int cg = col0 + nb + tn * 16 + n16;
      out[(size_t)r * CDIM + cg] = av[j] + bias[cg];
    }
  }
}

// ====================== Kernel 4: new_mask = sqrt(mask) ====================
__global__ __launch_bounds__(256) void sqrt_mask_kernel(
    const float* __restrict__ mask, float* __restrict__ out) {
  int i = blockIdx.x * blockDim.x + threadIdx.x;
  if (i < BATCH * SEQ) out[i] = sqrtf(mask[i]);
}

// ================================ Launch ===================================
extern "C" void kernel_launch(void* const* d_in, const int* in_sizes, int n_in,
                              void* d_out, int out_size, void* d_ws, size_t ws_size,
                              hipStream_t stream) {
  const float* h      = (const float*)d_in[0];
  const float* mask   = (const float*)d_in[1];
  const float* w_qkv  = (const float*)d_in[2];
  const float* w_proj = (const float*)d_in[3];
  const float* b_proj = (const float*)d_in[4];
  float* out = (float*)d_out;

  const size_t BHND = (size_t)BATCH * NHEADS * SEQ * HDIM;   // 4.2M elems
  __bf16* qkv  = (__bf16*)d_ws;
  __bf16* attn = (__bf16*)((char*)d_ws + 3 * BHND * sizeof(__bf16));

  // 1) QKV projection: one branch-free instantiation per segment
  qkv_gemm_kernel<0><<<dim3(8, 64), 256, 0, stream>>>(h, w_qkv,        qkv);
  qkv_gemm_kernel<1><<<dim3(8, 64), 256, 0, stream>>>(h, w_qkv + 1024, qkv + BHND);
  qkv_gemm_kernel<2><<<dim3(8, 64), 256, 0, stream>>>(h, w_qkv + 2048, qkv + 2 * BHND);
  // 2) attention: B*H*(N/128) = 512 blocks
  attn_kernel<<<dim3(512), 256, 0, stream>>>(qkv, mask, attn);
  // 3) output projection: grid (1024/128, 4096/64)
  proj_gemm_kernel<<<dim3(8, 64), 256, 0, stream>>>(attn, w_proj, b_proj, out);
  // 4) new_mask tail
  sqrt_mask_kernel<<<dim3(16), 256, 0, stream>>>(mask, out + (size_t)ROWS * CDIM);
}
